// TRCNB_4466765988305
// MI455X (gfx1250) — matrix operand order repaired and checked by reading, not compiled, not run on hardware
//
#include <hip/hip_runtime.h>

typedef _Float16 f16;
typedef __attribute__((ext_vector_type(16))) _Float16 v16h;
typedef __attribute__((ext_vector_type(8)))  float    v8f;

#define kH   128
#define kW   128
#define kHW  (kH * kW)          // 16384
#define kB   8
#define kNPIX (kB * kHW)        // 131072

// ---------------------------------------------------------------------------
// Pack OIHW f32 weights into WMMA A-fragment-ordered f16:
//   dst[((mt*KT + kt)*32 + lane)*16 + h]  with K global index = tap*Cin + ci
// A 16x32 f16 order used here (B matches): lane L -> M = L&15; half h -> K = (L>=16?16:0) + h
// ---------------------------------------------------------------------------
__global__ void k_pack_w(const float* __restrict__ w, f16* __restrict__ dst,
                         int Cout, int Cin, int KS) {
  int KT = (Cin * KS * KS) >> 5;
  int total = (Cout >> 4) * KT * 512;
  int idx = blockIdx.x * blockDim.x + threadIdx.x;
  if (idx >= total) return;
  int h    = idx & 15;
  int lane = (idx >> 4) & 31;
  int t    = idx >> 9;
  int kt   = t % KT;
  int mt   = t / KT;
  int klocal = ((lane >> 4) << 4) + h;
  int kg  = kt * 32 + klocal;
  int tap = kg / Cin;
  int ci  = kg - tap * Cin;
  int co  = mt * 16 + (lane & 15);
  dst[idx] = (f16)w[(size_t)(co * Cin + ci) * (KS * KS) + tap];
}

// ---------------------------------------------------------------------------
// Implicit-GEMM conv via v_wmma_f32_16x16x32_f16, fully unrolled over
// taps x k-chunks (KS, KCN compile-time). Each wave computes a
// 32(outch) x 16(pixel) tile: two accumulators share every B fragment.
//  * input: pixel-major f16 [ (b*HW+p)*inStride + c ]
//  * KS==3 -> reflect padding; KS==1 -> no padding
//  * epilogue: +bias, relu, +residual(NCHW f32), write NCHW f32 and/or PM f16
// ---------------------------------------------------------------------------
template <int KS, int KCN>
__global__ void k_conv_wmma(const f16* __restrict__ in_pm, int inStride,
                            const f16* __restrict__ wpack,
                            const float* __restrict__ bias,
                            int relu,
                            const float* __restrict__ res, int resStride,
                            float* __restrict__ outF, int outStride, int outCoff,
                            f16* __restrict__ outPm, int outPmStride, int outPmCoff) {
  constexpr int KT = KCN * KS * KS;
  int gtid = blockIdx.x * blockDim.x + threadIdx.x;
  int wave = gtid >> 5;
  int lane = gtid & 31;
  const int ntiles = kNPIX / 16;
  int mt2 = wave / ntiles;          // pair of 16-row M tiles (32 out channels)
  int nt  = wave - mt2 * ntiles;
  int n0 = nt * 16;
  int b  = n0 / kHW;
  int p0 = n0 - b * kHW;
  int y  = p0 / kW;
  int x0 = p0 - y * kW;
  int col  = lane & 15;
  int lx   = x0 + col;
  int koff = (lane >> 4) << 4;      // 0 or 16 (B fragment K offset per lane half)
  const v16h* wp0 = (const v16h*)wpack + ((size_t)(2 * mt2) * KT) * 32 + lane;
  const v16h* wp1 = wp0 + (size_t)KT * 32;
  v8f acc0 = {};
  v8f acc1 = {};
#pragma unroll
  for (int tap = 0; tap < KS * KS; ++tap) {
    int py = y, px = lx;
    if (KS == 3) {
      const int dy = tap / 3 - 1;
      const int dx = tap % 3 - 1;
      py = y + dy;  py = py < 0 ? -py : py;  py = py >= kH ? 2 * kH - 2 - py : py;
      px = lx + dx; px = px < 0 ? -px : px;  px = px >= kW ? 2 * kW - 2 - px : px;
    }
    const f16* bbase = in_pm + (size_t)(b * kHW + py * kW + px) * inStride + koff;
#pragma unroll
    for (int kc = 0; kc < KCN; ++kc) {
      const int kt = tap * KCN + kc;
      v16h bfrag = *(const v16h*)(bbase + kc * 32);
      v16h a0 = wp0[(size_t)kt * 32];
      v16h a1 = wp1[(size_t)kt * 32];
      acc0 = __builtin_amdgcn_wmma_f32_16x16x32_f16(false, a0, false, bfrag,
                                                    (short)0, acc0, false, false);
      acc1 = __builtin_amdgcn_wmma_f32_16x16x32_f16(false, a1, false, bfrag,
                                                    (short)0, acc1, false, false);
    }
  }
  int p = p0 + col;
  int half = (lane >> 4) * 8;       // C/D layout: vgpr r -> M = r + (lane>=16 ? 8 : 0)
#pragma unroll
  for (int part = 0; part < 2; ++part) {
    int cobase = mt2 * 32 + part * 16 + half;
#pragma unroll
    for (int r = 0; r < 8; ++r) {
      int co = cobase + r;
      float vv = part ? acc1[r] : acc0[r];
      if (bias) vv += bias[co];
      if (relu) vv = fmaxf(vv, 0.f);
      if (res)  vv += res[(size_t)(b * resStride + co) * kHW + p];
      if (outF) outF[(size_t)(b * outStride + outCoff + co) * kHW + p] = vv;
      if (outPm) outPm[(size_t)(b * kHW + p) * outPmStride + outPmCoff + co] = (f16)vv;
    }
  }
}

// NCHW f32 -> pixel-major f16
__global__ void k_to_pm(const float* __restrict__ src, f16* __restrict__ pm,
                        int Cc, int pmStride, int coff) {
  int i = blockIdx.x * blockDim.x + threadIdx.x;
  if (i >= kNPIX * Cc) return;
  int p = i % kHW;
  int c = (i / kHW) % Cc;
  int b = i / (kHW * Cc);
  pm[(size_t)(b * kHW + p) * pmStride + coff + c] = (f16)src[(size_t)(b * Cc + c) * kHW + p];
}

// per-pixel layernorm over 64 channels, optional f32 NCHW and/or PM f16 out
__global__ void k_layernorm(const float* __restrict__ x, const float* __restrict__ w,
                            const float* __restrict__ bsh,
                            f16* __restrict__ out_pm, float* __restrict__ out_f32) {
  int n = blockIdx.x * blockDim.x + threadIdx.x;
  if (n >= kNPIX) return;
  int b = n / kHW, p = n - b * kHW;
  const float* base = x + (size_t)b * 64 * kHW + p;
  float s = 0.f, s2 = 0.f;
  for (int c = 0; c < 64; ++c) { float v = base[(size_t)c * kHW]; s += v; s2 += v * v; }
  float mu  = s * (1.f / 64.f);
  float var = s2 * (1.f / 64.f) - mu * mu;
  float inv = rsqrtf(var + 1e-5f);
  for (int c = 0; c < 64; ++c) {
    float v = (base[(size_t)c * kHW] - mu) * inv * w[c] + bsh[c];
    if (out_pm)  out_pm[(size_t)n * 64 + c] = (f16)v;
    if (out_f32) out_f32[(size_t)(b * 64 + c) * kHW + p] = v;
  }
}

// depthwise 3x3, zero pad, 64ch, f32 NCHW in -> PM f16 out
__global__ void k_dwconv(const float* __restrict__ x, const float* __restrict__ w,
                         f16* __restrict__ out_pm) {
  int i = blockIdx.x * blockDim.x + threadIdx.x;
  if (i >= kNPIX * 64) return;
  int c = i & 63;
  int n = i >> 6;
  int b = n / kHW, p = n - b * kHW, y = p / kW, xx = p - y * kW;
  float s = 0.f;
  for (int t = 0; t < 9; ++t) {
    int dy = t / 3 - 1, dx = t % 3 - 1;
    int py = y + dy, px = xx + dx;
    if (py < 0 || py >= kH || px < 0 || px >= kW) continue;
    s += w[c * 9 + t] * x[(size_t)(b * 64 + c) * kHW + py * kW + px];
  }
  out_pm[(size_t)n * 64 + c] = (f16)s;
}

// per-row reduction over HW: mode 0 -> 1/max(||row||,1e-12) ; mode 1 -> mean
__global__ void k_row(const float* __restrict__ x, float* __restrict__ outv, int mode) {
  __shared__ float smem[256];
  size_t row = blockIdx.x;
  const float* base = x + row * kHW;
  float s = 0.f;
  for (int i = threadIdx.x; i < kHW; i += 256) { float v = base[i]; s += mode ? v : v * v; }
  smem[threadIdx.x] = s; __syncthreads();
  for (int o = 128; o > 0; o >>= 1) {
    if ((int)threadIdx.x < o) smem[threadIdx.x] += smem[threadIdx.x + o];
    __syncthreads();
  }
  if (threadIdx.x == 0) {
    if (mode) outv[row] = smem[0] * (1.f / (float)kHW);
    else { float nm = sqrtf(smem[0]); nm = nm > 1e-12f ? nm : 1e-12f; outv[row] = 1.f / nm; }
  }
}

// S[b,h,c,d] = <q_row, k_row> * (1/|q|)(1/|k|) * 8^-0.5
__global__ void k_qk(const float* __restrict__ q, const float* __restrict__ k,
                     const float* __restrict__ qn, const float* __restrict__ kn,
                     float* __restrict__ S) {
  int id = blockIdx.x;
  int d = id & 7, c = (id >> 3) & 7, h = (id >> 6) & 7, b = id >> 9;
  int rq = b * 64 + h * 8 + c;
  int rk = b * 64 + h * 8 + d;
  __shared__ float smem[256];
  const float* qb = q + (size_t)rq * kHW;
  const float* kb = k + (size_t)rk * kHW;
  float s = 0.f;
  for (int i = threadIdx.x; i < kHW; i += 256) s += qb[i] * kb[i];
  smem[threadIdx.x] = s; __syncthreads();
  for (int o = 128; o > 0; o >>= 1) {
    if ((int)threadIdx.x < o) smem[threadIdx.x] += smem[threadIdx.x + o];
    __syncthreads();
  }
  if (threadIdx.x == 0) S[id] = smem[0] * qn[rq] * kn[rk] * 0.35355339059327373f;
}

__global__ void k_softmax8(const float* __restrict__ S, float* __restrict__ A) {
  int i = blockIdx.x * blockDim.x + threadIdx.x;
  if (i >= 512) return;
  const float* r = S + i * 8;
  float m = r[0];
  for (int j = 1; j < 8; ++j) m = fmaxf(m, r[j]);
  float e[8], s = 0.f;
  for (int j = 0; j < 8; ++j) { e[j] = expf(r[j] - m); s += e[j]; }
  float inv = 1.f / s;
  for (int j = 0; j < 8; ++j) A[i * 8 + j] = e[j] * inv;
}

// out[c,e] = sum_d A[c,d] * vhat[d,e] -> PM f16
__global__ void k_attn_out(const float* __restrict__ A, const float* __restrict__ v,
                           const float* __restrict__ vn, f16* __restrict__ out_pm) {
  int i = blockIdx.x * blockDim.x + threadIdx.x;
  if (i >= kNPIX * 64) return;
  int e = i % kHW;
  int c = (i / kHW) & 63;
  int b = i / (kHW * 64);
  int h = c >> 3;
  const float* a = A + (((b * 8 + h) * 8 + (c & 7)) * 8);
  float s = 0.f;
  for (int d = 0; d < 8; ++d) {
    int rv = b * 64 + h * 8 + d;
    s += a[d] * v[(size_t)rv * kHW + e] * vn[rv];
  }
  out_pm[(size_t)(b * kHW + e) * 64 + c] = (f16)s;
}

// grouped 3x3 conv (groups=64, Cin/grp=4, Cout/grp=2), reflect pad
__global__ void k_mlp_dw(const float* __restrict__ y1, const float* __restrict__ w,
                         float* __restrict__ y2) {
  int i = blockIdx.x * blockDim.x + threadIdx.x;
  if (i >= kNPIX * 128) return;
  int p  = i % kHW;
  int co = (i / kHW) & 127;
  int b  = i / (kHW * 128);
  int grp = co >> 1;
  int yy = p / kW, xx = p - (p / kW) * kW;
  float s = 0.f;
  for (int ci = 0; ci < 4; ++ci) {
    const float* base = y1 + (size_t)(b * 256 + grp * 4 + ci) * kHW;
    const float* wb = w + (size_t)(co * 4 + ci) * 9;
    for (int t = 0; t < 9; ++t) {
      int dy = t / 3 - 1, dx = t % 3 - 1;
      int py = yy + dy; py = py < 0 ? -py : py; py = py >= kH ? 2 * kH - 2 - py : py;
      int px = xx + dx; px = px < 0 ? -px : px; px = px >= kW ? 2 * kW - 2 - px : px;
      s += wb[t] * base[py * kW + px];
    }
  }
  y2[(size_t)(b * 128 + co) * kHW + p] = s;
}

// silu gate + batchnorm -> PM f16
__global__ void k_mlp_gate(const float* __restrict__ y2, const float* __restrict__ g,
                           const float* __restrict__ bb, const float* __restrict__ rm,
                           const float* __restrict__ rv, f16* __restrict__ out_pm) {
  int i = blockIdx.x * blockDim.x + threadIdx.x;
  if (i >= kNPIX * 64) return;
  int p = i % kHW;
  int c = (i / kHW) & 63;
  int b = i / (kHW * 64);
  float a = y2[(size_t)(b * 128 + c) * kHW + p];
  float m = y2[(size_t)(b * 128 + 64 + c) * kHW + p];
  float y = (a / (1.f + expf(-a))) * m;
  y = (y - rm[c]) * rsqrtf(rv[c] + 1e-5f) * g[c] + bb[c];
  out_pm[(size_t)(b * kHW + p) * 64 + c] = (f16)y;
}

// pm[..] = f16(a + b), a has aStride channels, b is 64-ch NCHW
__global__ void k_add_pm(const float* __restrict__ a, int aStride,
                         const float* __restrict__ b2, f16* __restrict__ pm,
                         int pmStride, int coff) {
  int i = blockIdx.x * blockDim.x + threadIdx.x;
  if (i >= kNPIX * 64) return;
  int p = i % kHW;
  int c = (i / kHW) & 63;
  int b = i / (kHW * 64);
  float v = a[(size_t)(b * aStride + c) * kHW + p] + b2[(size_t)(b * 64 + c) * kHW + p];
  pm[(size_t)(b * kHW + p) * pmStride + coff + c] = (f16)v;
}

// xe -> dense (f32 128-stride ch 0..63) + dense PM f16
__global__ void k_copy_dense(const float* __restrict__ xe, float* __restrict__ df,
                             f16* __restrict__ dpm) {
  int i = blockIdx.x * blockDim.x + threadIdx.x;
  if (i >= kNPIX * 64) return;
  int p = i % kHW;
  int c = (i / kHW) & 63;
  int b = i / (kHW * 64);
  float v = xe[(size_t)(b * 64 + c) * kHW + p];
  df[(size_t)(b * 128 + c) * kHW + p] = v;
  dpm[(size_t)(b * kHW + p) * 128 + c] = (f16)v;
}

// channel-attention pooled means: xh[bg,c,y] over x, xw[bg,c,x] over y
__global__ void k_ca_pool(const float* __restrict__ df, float* __restrict__ xh,
                          float* __restrict__ xw) {
  int i = blockIdx.x * blockDim.x + threadIdx.x;
  const int tot = 256 * 4 * 128;
  if (i >= 2 * tot) return;
  int which = i / tot;
  int r = i - which * tot;
  int idx = r & 127;
  int c = (r >> 7) & 3;
  int bg = r >> 9;
  int b = bg >> 5;
  int gch = ((bg & 31) << 2) + c;
  const float* base = df + (size_t)(b * 128 + gch) * kHW;
  float s = 0.f;
  if (which == 0) { for (int xx = 0; xx < 128; ++xx) s += base[idx * kW + xx]; xh[r] = s * (1.f / 128.f); }
  else            { for (int yy = 0; yy < 128; ++yy) s += base[yy * kW + idx]; xw[r] = s * (1.f / 128.f); }
}

// 1x1 conv (4->4) + bias + sigmoid on pooled vectors
__global__ void k_ca1(const float* __restrict__ xh, const float* __restrict__ xw,
                      const float* __restrict__ w, const float* __restrict__ bsh,
                      float* __restrict__ gh, float* __restrict__ gw) {
  int i = blockIdx.x * blockDim.x + threadIdx.x;
  const int tot = 256 * 4 * 128;
  if (i >= 2 * tot) return;
  int which = i / tot;
  int r = i - which * tot;
  int idx = r & 127;
  int co = (r >> 7) & 3;
  int bg = r >> 9;
  const float* src = which ? xw : xh;
  float s = bsh[co];
  for (int ci = 0; ci < 4; ++ci) s += w[co * 4 + ci] * src[(bg * 4 + ci) * 128 + idx];
  float sg = 1.f / (1.f + expf(-s));
  (which ? gw : gh)[r] = sg;
}

// t = gx * sig(xh[y]) * sig(xw[x]); block partial sums (sum, sumsq) per bg chunk
__global__ void k_ca_gate(const float* __restrict__ df, const float* __restrict__ gh,
                          const float* __restrict__ gw, float* __restrict__ t,
                          float* __restrict__ pA, float* __restrict__ pB) {
  int i = blockIdx.x * blockDim.x + threadIdx.x;
  int p = i & (kHW - 1);
  int c = (i >> 14) & 3;
  int bg = i >> 16;
  int b = bg >> 5;
  int gch = ((bg & 31) << 2) + c;
  int yy = p >> 7, xx = p & 127;
  float v = df[(size_t)(b * 128 + gch) * kHW + p] *
            gh[(bg * 4 + c) * 128 + yy] * gw[(bg * 4 + c) * 128 + xx];
  t[i] = v;
  __shared__ float s1[256], s2[256];
  s1[threadIdx.x] = v; s2[threadIdx.x] = v * v; __syncthreads();
  for (int o = 128; o > 0; o >>= 1) {
    if ((int)threadIdx.x < o) { s1[threadIdx.x] += s1[threadIdx.x + o]; s2[threadIdx.x] += s2[threadIdx.x + o]; }
    __syncthreads();
  }
  if (threadIdx.x == 0) { pA[blockIdx.x] = s1[0]; pB[blockIdx.x] = s2[0]; }
}

__global__ void k_reduce_part(const float* __restrict__ part, float* __restrict__ outv,
                              int rows, int per) {
  int r = blockIdx.x * blockDim.x + threadIdx.x;
  if (r >= rows) return;
  const float* pp = part + (size_t)r * per;
  float s = 0.f;
  for (int i = 0; i < per; ++i) s += pp[i];
  outv[r] = s;
}

// groupnorm over (4,H,W) per bg; in-place; partial sums per (bg,c) chunk
__global__ void k_ca_gn(float* __restrict__ t, const float* __restrict__ gsum,
                        const float* __restrict__ gsum2, const float* __restrict__ gng,
                        const float* __restrict__ gnb, float* __restrict__ pA) {
  int i = blockIdx.x * blockDim.x + threadIdx.x;
  int c = (i >> 14) & 3;
  int bg = i >> 16;
  float mu  = gsum[bg] * (1.f / 65536.f);
  float var = gsum2[bg] * (1.f / 65536.f) - mu * mu;
  float v = (t[i] - mu) * rsqrtf(var + 1e-5f) * gng[c] + gnb[c];
  t[i] = v;
  __shared__ float smem[256];
  smem[threadIdx.x] = v; __syncthreads();
  for (int o = 128; o > 0; o >>= 1) {
    if ((int)threadIdx.x < o) smem[threadIdx.x] += smem[threadIdx.x + o];
    __syncthreads();
  }
  if (threadIdx.x == 0) pA[blockIdx.x] = smem[0];
}

// ca3: 3x3 zero-pad 4->4 conv per group; partial sums per (bg,c) chunk
__global__ void k_ca3(const float* __restrict__ df, const float* __restrict__ w,
                      const float* __restrict__ bsh, float* __restrict__ x2,
                      float* __restrict__ pA) {
  int i = blockIdx.x * blockDim.x + threadIdx.x;
  int p = i & (kHW - 1);
  int c = (i >> 14) & 3;
  int bg = i >> 16;
  int b = bg >> 5;
  int g0 = (bg & 31) << 2;
  int yy = p >> 7, xx = p & 127;
  float s = bsh[c];
  for (int ci = 0; ci < 4; ++ci) {
    const float* base = df + (size_t)(b * 128 + g0 + ci) * kHW;
    for (int t = 0; t < 9; ++t) {
      int dy = t / 3 - 1, dx = t % 3 - 1;
      int py = yy + dy, px = xx + dx;
      if (py < 0 || py >= kH || px < 0 || px >= kW) continue;
      s += w[(c * 4 + ci) * 9 + t] * base[py * kW + px];
    }
  }
  x2[i] = s;
  __shared__ float smem[256];
  smem[threadIdx.x] = s; __syncthreads();
  for (int o = 128; o > 0; o >>= 1) {
    if ((int)threadIdx.x < o) smem[threadIdx.x] += smem[threadIdx.x + o];
    __syncthreads();
  }
  if (threadIdx.x == 0) pA[blockIdx.x] = smem[0];
}

__global__ void k_softmax4(const float* __restrict__ x1m, const float* __restrict__ x2m,
                           float* __restrict__ x11, float* __restrict__ x21) {
  int bg = blockIdx.x * blockDim.x + threadIdx.x;
  if (bg >= 256) return;
  for (int which = 0; which < 2; ++which) {
    const float* src = which ? x2m : x1m;
    float* dst = which ? x21 : x11;
    float a[4], mx = -1e30f;
    for (int c = 0; c < 4; ++c) { a[c] = src[bg * 4 + c] * (1.f / (float)kHW); mx = fmaxf(mx, a[c]); }
    float s = 0.f;
    for (int c = 0; c < 4; ++c) { a[c] = expf(a[c] - mx); s += a[c]; }
    float inv = 1.f / s;
    for (int c = 0; c < 4; ++c) dst[bg * 4 + c] = a[c] * inv;
  }
}

// squeeze-excite over dense means: s = sigmoid(W2 relu(W1 mean + b1) + b2)
__global__ void k_chsp_s(const float* __restrict__ dmean, const float* __restrict__ w1,
                         const float* __restrict__ b1, const float* __restrict__ w2,
                         const float* __restrict__ b2, float* __restrict__ sv) {
  __shared__ float hid[32];
  int b = blockIdx.x;
  int t = threadIdx.x;
  if (t < 32) {
    float s = b1[t];
    for (int ci = 0; ci < 128; ++ci) s += w1[t * 128 + ci] * dmean[b * 128 + ci];
    hid[t] = fmaxf(s, 0.f);
  }
  __syncthreads();
  float s = b2[t];
  for (int ci = 0; ci < 32; ++ci) s += w2[t * 32 + ci] * hid[ci];
  sv[b * 128 + t] = 1.f / (1.f + expf(-s));
}

// weights = x11@x2 + x21@x1 ; ch = 0.8*gx*sig(w)+0.2*gx ; + gx*(0.4+0.6*s) -> PM f16
__global__ void k_chsp_final(const float* __restrict__ df, const float* __restrict__ t,
                             const float* __restrict__ x2, const float* __restrict__ x11,
                             const float* __restrict__ x21, const float* __restrict__ sv,
                             f16* __restrict__ chsp_pm) {
  int i = blockIdx.x * blockDim.x + threadIdx.x;
  int p = i & (kHW - 1);
  int c4 = (i >> 14) & 3;
  int bg = i >> 16;
  int b = bg >> 5;
  int gch = ((bg & 31) << 2) + c4;
  float wsum = 0.f;
  for (int c = 0; c < 4; ++c)
    wsum += x11[bg * 4 + c] * x2[((size_t)(bg * 4 + c) << 14) + p] +
            x21[bg * 4 + c] * t [((size_t)(bg * 4 + c) << 14) + p];
  float gx = df[(size_t)(b * 128 + gch) * kHW + p];
  float sig = 1.f / (1.f + expf(-wsum));
  float ch = 0.8f * gx * sig + 0.2f * gx;
  float outv = ch + gx * (0.4f + 0.6f * sv[b * 128 + gch]);
  chsp_pm[(size_t)(b * kHW + p) * 128 + gch] = (f16)outv;
}

// ---------------------------------------------------------------------------
extern "C" void kernel_launch(void* const* d_in, const int* in_sizes, int n_in,
                              void* d_out, int out_size, void* d_ws, size_t ws_size,
                              hipStream_t stream) {
  (void)in_sizes; (void)n_in; (void)out_size; (void)ws_size;
  const float* x         = (const float*)d_in[0];
  const float* embed_w   = (const float*)d_in[1];
  const float* n1_w      = (const float*)d_in[2];
  const float* n1_b      = (const float*)d_in[3];
  const float* attn_dw   = (const float*)d_in[4];
  const float* attn_q    = (const float*)d_in[5];
  const float* attn_k    = (const float*)d_in[6];
  const float* attn_v    = (const float*)d_in[7];
  const float* attn_proj = (const float*)d_in[8];
  const float* n2_w      = (const float*)d_in[9];
  const float* n2_b      = (const float*)d_in[10];
  const float* mlp_in_w  = (const float*)d_in[11];
  const float* mlp_dw_w  = (const float*)d_in[12];
  const float* bn_g      = (const float*)d_in[13];
  const float* bn_b      = (const float*)d_in[14];
  const float* bn_rm     = (const float*)d_in[15];
  const float* bn_rv     = (const float*)d_in[16];
  const float* mlp_out_w = (const float*)d_in[17];
  const float* ffn_w     = (const float*)d_in[18];

  char* ws = (char*)d_ws;
  size_t off = 0;
  auto alloc = [&](size_t bytes) -> void* {
    void* p = ws + off;
    off += (bytes + 255) & ~(size_t)255;
    return p;
  };
  f16*   x_pm   = (f16*)  alloc((size_t)kNPIX * 32 * 2);
  float* xe     = (float*)alloc((size_t)kNPIX * 64 * 4);
  float* lnf    = (float*)alloc((size_t)kNPIX * 64 * 4);
  f16*   pmA    = (f16*)  alloc((size_t)kNPIX * 64 * 2);
  float* qb     = (float*)alloc((size_t)kNPIX * 64 * 4);
  float* kb     = (float*)alloc((size_t)kNPIX * 64 * 4);
  float* vbuf   = (float*)alloc((size_t)kNPIX * 64 * 4);
  f16*   pmB    = (f16*)  alloc((size_t)kNPIX * 64 * 2);
  float* gbuf   = (float*)alloc((size_t)kNPIX * 64 * 4);
  float* y1     = (float*)alloc((size_t)kNPIX * 256 * 4);
  float* y2     = (float*)alloc((size_t)kNPIX * 128 * 4);
  float* densF  = (float*)alloc((size_t)kNPIX * 128 * 4);
  f16*   densPm = (f16*)  alloc((size_t)kNPIX * 128 * 2);
  f16*   wpack  = (f16*)  alloc((size_t)1 << 20);
  float* sm     = (float*)alloc((size_t)4 << 20);
  // aliases onto dead regions
  float* caT    = y1;                         // (bg,4,HW) after mlp is done
  float* caX2   = y1 + (size_t)kNPIX * 128;
  f16*   chspPm = (f16*)y2;
  f16*   catPm  = (f16*)kb;
  float* outp   = (float*)d_out;

  float* qn = sm, *kn = sm + 512, *vn = sm + 1024;
  float* S = sm + 1536;      float* A = sm + 5632;
  float* gsum = sm + 9728;   float* gsum2 = sm + 9984;
  float* x1mean = sm + 10240, *x2mean = sm + 11264;
  float* x11 = sm + 12288, *x21 = sm + 13312;
  float* dmean = sm + 14336, *sv = sm + 15360;
  float* xh = sm + 16384, *xw = sm + 147456;
  float* gh = sm + 278528, *gw = sm + 409600;
  float* pA = sm + 540672, *pB = sm + 606208;

  auto conv = [&](const f16* inpm, int inStride, int Cin,
                  const float* wsrc, const float* bias,
                  int Cout, int KS, int relu,
                  const float* res, int resStride,
                  float* outF, int outStride, int outCoff,
                  f16* outPm, int outPmStride, int outPmCoff) {
    int KT = (Cin * KS * KS) >> 5;
    int ptotal = (Cout >> 4) * KT * 512;
    k_pack_w<<<dim3((ptotal + 255) / 256), dim3(256), 0, stream>>>(wsrc, wpack, Cout, Cin, KS);
    int waves = (Cout >> 5) * (kNPIX / 16);   // one wave = 32 outch x 16 pixels
    dim3 grid(waves / 8), blk(256);
    if (KS == 1 && Cin == 64)
      k_conv_wmma<1, 2><<<grid, blk, 0, stream>>>(inpm, inStride, wpack, bias, relu,
          res, resStride, outF, outStride, outCoff, outPm, outPmStride, outPmCoff);
    else if (KS == 1 && Cin == 128)
      k_conv_wmma<1, 4><<<grid, blk, 0, stream>>>(inpm, inStride, wpack, bias, relu,
          res, resStride, outF, outStride, outCoff, outPm, outPmStride, outPmCoff);
    else if (KS == 3 && Cin == 32)
      k_conv_wmma<3, 1><<<grid, blk, 0, stream>>>(inpm, inStride, wpack, bias, relu,
          res, resStride, outF, outStride, outCoff, outPm, outPmStride, outPmCoff);
    else if (KS == 3 && Cin == 64)
      k_conv_wmma<3, 2><<<grid, blk, 0, stream>>>(inpm, inStride, wpack, bias, relu,
          res, resStride, outF, outStride, outCoff, outPm, outPmStride, outPmCoff);
    else if (KS == 3 && Cin == 96)
      k_conv_wmma<3, 3><<<grid, blk, 0, stream>>>(inpm, inStride, wpack, bias, relu,
          res, resStride, outF, outStride, outCoff, outPm, outPmStride, outPmCoff);
    else if (KS == 3 && Cin == 128)
      k_conv_wmma<3, 4><<<grid, blk, 0, stream>>>(inpm, inStride, wpack, bias, relu,
          res, resStride, outF, outStride, outCoff, outPm, outPmStride, outPmCoff);
  };

  // ---- embed ----
  k_to_pm<<<dim3(kNPIX * 32 / 256), 256, 0, stream>>>(x, x_pm, 32, 32, 0);
  conv(x_pm, 32, 32, embed_w, nullptr, 64, 3, 0, nullptr, 0, xe, 64, 0, nullptr, 0, 0);

  // ---- attention branch ----
  k_layernorm<<<dim3(kNPIX / 256), 256, 0, stream>>>(xe, n1_w, n1_b, nullptr, lnf);
  k_dwconv<<<dim3(kNPIX * 64 / 256), 256, 0, stream>>>(lnf, attn_dw, pmA);
  conv(pmA, 64, 64, attn_q, nullptr, 64, 1, 0, nullptr, 0, qb,   64, 0, nullptr, 0, 0);
  conv(pmA, 64, 64, attn_k, nullptr, 64, 1, 0, nullptr, 0, kb,   64, 0, nullptr, 0, 0);
  conv(pmA, 64, 64, attn_v, nullptr, 64, 1, 0, nullptr, 0, vbuf, 64, 0, nullptr, 0, 0);
  k_row<<<dim3(512), 256, 0, stream>>>(qb, qn, 0);
  k_row<<<dim3(512), 256, 0, stream>>>(kb, kn, 0);
  k_row<<<dim3(512), 256, 0, stream>>>(vbuf, vn, 0);
  k_qk<<<dim3(4096), 256, 0, stream>>>(qb, kb, qn, kn, S);
  k_softmax8<<<dim3(2), 256, 0, stream>>>(S, A);
  k_attn_out<<<dim3(kNPIX * 64 / 256), 256, 0, stream>>>(A, vbuf, vn, pmB);
  conv(pmB, 64, 64, attn_proj, nullptr, 64, 1, 0, xe, 64, gbuf, 64, 0, nullptr, 0, 0);

  // ---- mlp branch ----
  k_layernorm<<<dim3(kNPIX / 256), 256, 0, stream>>>(gbuf, n2_w, n2_b, pmA, nullptr);
  conv(pmA, 64, 64, mlp_in_w, nullptr, 256, 1, 0, nullptr, 0, y1, 256, 0, nullptr, 0, 0);
  k_mlp_dw<<<dim3(kNPIX * 128 / 256), 256, 0, stream>>>(y1, mlp_dw_w, y2);
  k_mlp_gate<<<dim3(kNPIX * 64 / 256), 256, 0, stream>>>(y2, bn_g, bn_b, bn_rm, bn_rv, pmB);
  conv(pmB, 64, 64, mlp_out_w, nullptr, 64, 1, 0, gbuf, 64, gbuf, 64, 0, nullptr, 0, 0);

  // x1 = g + xe
  k_add_pm<<<dim3(kNPIX * 64 / 256), 256, 0, stream>>>(gbuf, 64, xe, catPm, 128, 0);

  // ---- dense blocks ----
  k_copy_dense<<<dim3(kNPIX * 64 / 256), 256, 0, stream>>>(xe, densF, densPm);
  for (int blk = 0; blk < 3; ++blk) {
    const float* c1_w  = (const float*)d_in[19 + blk * 16 + 0];
    const float* c1_b  = (const float*)d_in[19 + blk * 16 + 1];
    const float* c2_w  = (const float*)d_in[19 + blk * 16 + 2];
    const float* c2_b  = (const float*)d_in[19 + blk * 16 + 3];
    const float* ca1_w = (const float*)d_in[19 + blk * 16 + 4];
    const float* ca1_b = (const float*)d_in[19 + blk * 16 + 5];
    const float* ca3_w = (const float*)d_in[19 + blk * 16 + 6];
    const float* ca3_b = (const float*)d_in[19 + blk * 16 + 7];
    const float* gn_g  = (const float*)d_in[19 + blk * 16 + 8];
    const float* gn_b  = (const float*)d_in[19 + blk * 16 + 9];
    const float* s1_w  = (const float*)d_in[19 + blk * 16 + 10];
    const float* s1_b  = (const float*)d_in[19 + blk * 16 + 11];
    const float* s2_w  = (const float*)d_in[19 + blk * 16 + 12];
    const float* s2_b  = (const float*)d_in[19 + blk * 16 + 13];
    const float* fus_w = (const float*)d_in[19 + blk * 16 + 14];
    const float* fus_b = (const float*)d_in[19 + blk * 16 + 15];

    conv(densPm, 128, 64, c1_w, c1_b, 32, 3, 1, nullptr, 0, densF, 128, 64, densPm, 128, 64);
    conv(densPm, 128, 96, c2_w, c2_b, 32, 3, 1, nullptr, 0, densF, 128, 96, densPm, 128, 96);

    k_ca_pool<<<dim3(1024), 256, 0, stream>>>(densF, xh, xw);
    k_ca1<<<dim3(1024), 256, 0, stream>>>(xh, xw, ca1_w, ca1_b, gh, gw);
    k_ca_gate<<<dim3(kNPIX * 128 / 256), 256, 0, stream>>>(densF, gh, gw, caT, pA, pB);
    k_reduce_part<<<dim3(1), 256, 0, stream>>>(pA, gsum, 256, 256);
    k_reduce_part<<<dim3(1), 256, 0, stream>>>(pB, gsum2, 256, 256);
    k_ca_gn<<<dim3(kNPIX * 128 / 256), 256, 0, stream>>>(caT, gsum, gsum2, gn_g, gn_b, pA);
    k_reduce_part<<<dim3(4), 256, 0, stream>>>(pA, x1mean, 1024, 64);
    k_ca3<<<dim3(kNPIX * 128 / 256), 256, 0, stream>>>(densF, ca3_w, ca3_b, caX2, pA);
    k_reduce_part<<<dim3(4), 256, 0, stream>>>(pA, x2mean, 1024, 64);
    k_softmax4<<<dim3(1), 256, 0, stream>>>(x1mean, x2mean, x11, x21);
    k_row<<<dim3(1024), 256, 0, stream>>>(densF, dmean, 1);
    k_chsp_s<<<dim3(8), 128, 0, stream>>>(dmean, s1_w, s1_b, s2_w, s2_b, sv);
    k_chsp_final<<<dim3(kNPIX * 128 / 256), 256, 0, stream>>>(densF, caT, caX2, x11, x21, sv, chspPm);

    conv(chspPm, 128, 128, fus_w, fus_b, 64, 1, 0, densF, 128, densF, 128, 0, densPm, 128, 0);
  }

  // x2 = l + xe ; final ffn conv + residual xe -> d_out
  k_add_pm<<<dim3(kNPIX * 64 / 256), 256, 0, stream>>>(densF, 128, xe, catPm, 128, 64);
  conv(catPm, 128, 128, ffn_w, nullptr, 64, 3, 0, xe, 64, outp, 64, 0, nullptr, 0, 0);
}
